// CPADConvOffsetStage_88708254531618
// MI455X (gfx1250) — compile-verified, matrix operand order checked by reading, not run on hardware
//
#include <hip/hip_runtime.h>

typedef __attribute__((ext_vector_type(16))) _Float16 v16h;
typedef __attribute__((ext_vector_type(8)))  float    v8f;

#define BQ    8
#define HQ    128
#define WQ    128
#define CIN   64
#define COUT  64
#define NFLAT 576     // COUT * 3*3
#define HID   32
#define PC    16
#define GP    16

// ---------------------------------------------------------------------------
// Kernel 1: x_adapt = w_ca @ x   (1x1 conv 64->64). One block per (b, y) row.
// ---------------------------------------------------------------------------
__global__ __launch_bounds__(256) void ca_kernel(const float* __restrict__ x,
                                                 const float* __restrict__ wca,
                                                 float* __restrict__ xad) {
    __shared__ float xs[CIN * WQ];     // 32 KB: one row, all in-channels
    __shared__ float ws[COUT * CIN];   // 16 KB
    const int t  = threadIdx.x;
    const int wg = blockIdx.x;         // b*128 + y
    const int b  = wg >> 7;
    const int y  = wg & 127;

    for (int i = t; i < CIN * WQ; i += 256) {
        int c = i >> 7, p = i & 127;
        xs[i] = x[(((size_t)b * CIN + c) * HQ + y) * WQ + p];
    }
    for (int i = t; i < COUT * CIN; i += 256) ws[i] = wca[i];
    __syncthreads();

    for (int rep = 0; rep < 32; ++rep) {
        int idx = rep * 256 + t;
        int o = idx >> 7, p = idx & 127;   // wave-uniform o, coalesced p
        float acc = 0.f;
#pragma unroll 8
        for (int c = 0; c < CIN; ++c)
            acc += ws[o * CIN + c] * xs[c * WQ + p];
        xad[(((size_t)b * COUT + o) * HQ + y) * WQ + p] = acc;
    }
}

// ---------------------------------------------------------------------------
// Kernel 2: fused grid-sample -> MLP -> WMMA weight-gen -> deformable 3x3 mix.
// One block = 64 pixels of one row (4 waves x 16 pixels). 128 threads.
// ---------------------------------------------------------------------------
__global__ __launch_bounds__(128) void cpad_kernel(const float* __restrict__ offset,
                                                   const float* __restrict__ posi,
                                                   const float* __restrict__ w1,
                                                   const float* __restrict__ b1,
                                                   const float* __restrict__ w2,
                                                   const float* __restrict__ b2,
                                                   const float* __restrict__ bias,
                                                   const float* __restrict__ xad,
                                                   float* __restrict__ out) {
    __shared__ _Float16 w2h[NFLAT * HID];    // 36864 B, f16 B-matrix for WMMA
    __shared__ float    w1s[HID * PC];       // 2048 B
    __shared__ float    b1s[HID];            // 128 B
    __shared__ float    b2s[NFLAT];          // 2304 B
    __shared__ float    bs[COUT];            // 256 B
    __shared__ float    outb[64 * 65];       // 16640 B, stride 65 vs bank conflicts
    __shared__ int      metas[NFLAT];        // 2304 B: packed (o, ty, tx) per flat idx

    const int t    = threadIdx.x;
    const int lane = t & 31;
    const int wv   = t >> 5;
    const int wg   = blockIdx.x;             // b*256 + y*2 + half
    const int b    = wg >> 8;
    const int y    = (wg >> 1) & 127;
    const int xbase = (wg & 1) * 64;

    // ---- stage constants + tap metadata in LDS ----
    for (int i = t; i < NFLAT * HID; i += 128) w2h[i] = (_Float16)w2[i];
    for (int i = t; i < HID * PC;   i += 128) w1s[i] = w1[i];
    if (t < HID)  b1s[t] = b1[t];
    for (int i = t; i < NFLAT; i += 128) {
        b2s[i] = b2[i];
        int o  = i / 9;
        int tt = i - o * 9;
        metas[i] = (o << 4) | ((tt - (tt / 3) * 3) << 2) | (tt / 3);
    }
    if (t < COUT) bs[t] = bias[t];
    for (int i = t; i < 64 * 65; i += 128) outb[i] = 0.f;
    __syncthreads();

    // ---- grid-sample of posi_map (each lane: pixel = x0 + (lane&15)) ----
    const int x0 = xbase + wv * 16;
    const int px = x0 + (lane & 15);
    float dx = offset[(((size_t)b * 2 + 0) * HQ + y) * WQ + px] * (1.0f / 127.0f);
    float dy = offset[(((size_t)b * 2 + 1) * HQ + y) * WQ + px] * (1.0f / 127.0f);
    float gx = -1.f + (2.f / 127.f) * (float)px + dx;
    float gy = -1.f + (2.f / 127.f) * (float)y  + dy;
    float ix = fminf(fmaxf((gx + 1.f) * 0.5f * 15.f, 0.f), 15.f);
    float iy = fminf(fmaxf((gy + 1.f) * 0.5f * 15.f, 0.f), 15.f);
    int xi0 = (int)ix;                       // ix >= 0 -> trunc == floor
    int yi0 = (int)iy;
    int xi1 = min(xi0 + 1, GP - 1);
    int yi1 = min(yi0 + 1, GP - 1);
    float fx = ix - (float)xi0, fy = iy - (float)yi0;

    float up[PC];
#pragma unroll
    for (int c = 0; c < PC; ++c) {
        const float* pb = posi + c * (GP * GP);
        float v00 = pb[yi0 * GP + xi0];
        float v01 = pb[yi0 * GP + xi1];
        float v10 = pb[yi1 * GP + xi0];
        float v11 = pb[yi1 * GP + xi1];
        float top = v00 + (v01 - v00) * fx;
        float bot = v10 + (v11 - v10) * fx;
        up[c] = top + (bot - top) * fy;
    }

    // ---- h = relu(w1 @ up + b1); lower half computes h[0..15], upper h[16..31]
    float hj[16];
    const int hb = (lane >> 4) * 16;
#pragma unroll
    for (int j = 0; j < 16; ++j) {
        float acc = b1s[hb + j];
#pragma unroll
        for (int c = 0; c < PC; ++c) acc += w1s[(hb + j) * PC + c] * up[c];
        hj[j] = fmaxf(acc, 0.f);
    }

    // ---- assemble A fragment (16 px x 32 K, f16) per ISA 16-bit A layout:
    // lanes 0-15: K0-7 (elems 0-7), K16-23 (elems 8-15)
    // lanes 16-31: K8-15 (elems 0-7), K24-31 (elems 8-15)
    const bool lower = (lane < 16);
    v16h afrag;
#pragma unroll
    for (int j = 0; j < 8; ++j) {
        float send  = lower ? hj[8 + j] : hj[j];     // value the partner needs
        float other = __shfl_xor(send, 16, 32);
        float lo = lower ? hj[j]  : other;           // elems 0-7
        float hi = lower ? other  : hj[8 + j];       // elems 8-15
        afrag[j]     = (_Float16)lo;
        afrag[8 + j] = (_Float16)hi;
    }

    // ---- 36 WMMA tiles: weights[16px x 576] = A(16x32) x w2^T(32x576),
    // fused with deformable 3x3 tap accumulation (branch-free, clamped loads).
    const int kb    = (lane >> 4) * 8;    // K halfword base for B fragment
    const int nrow  = (lane >> 4) * 8;    // M base of this lane's C rows
    const int obase0 = (wv * 16 + nrow) * 65;
    const unsigned planeB = (unsigned)(b * COUT) << 14;   // b plane base (elems)
    const int xs0base = x0 + nrow;

#pragma unroll 2
    for (int n = 0; n < 36; ++n) {
        const int f = n * 16 + (lane & 15);          // flat weight index = o*9+t
        // B fragment: two 16B LDS loads (aligned: f*64B + {0,16} + {0,32})
        union { uint4 u[2]; v16h v; } bb;
        const uint4* wp = (const uint4*)(w2h + f * HID + kb);
        bb.u[0] = wp[0];
        bb.u[1] = wp[2];                             // +16 halfwords
        v8f c = {};
        c = __builtin_amdgcn_wmma_f32_16x16x32_f16(false, afrag, false, bb.v,
                                                   (short)0, c, false, false);
        const int m  = metas[f];
        const int o  = m >> 4;
        const int tx = ((m >> 2) & 3) - 1;
        const int ty = (m & 3) - 1;
        const float wb = b2s[f];

        const int  yy    = y + ty;
        const bool ymask = (yy >= 0) && (yy < HQ);
        const int  yyc   = min(max(yy, 0), HQ - 1);
        const unsigned rowbase = planeB + ((unsigned)o << 14) + ((unsigned)yyc << 7);
        const int  xs0 = xs0base + tx;

        // 8 unconditional clamped loads (single clause). Multiplicative masking
        // keeps the loads unconditionally live so the compiler cannot sink them
        // back under divergent branches (v*0 is not foldable without fast-math).
        float pv[8];
#pragma unroll
        for (int g = 0; g < 8; ++g) {
            int xx  = xs0 + g;
            int xxc = min(max(xx, 0), WQ - 1);
            float v   = xad[rowbase + (unsigned)xxc];
            float msk = (ymask && (xx == xxc)) ? 1.0f : 0.0f;
            pv[g] = v * msk;
        }
        const int ob = obase0 + o;
#pragma unroll
        for (int g = 0; g < 8; ++g) {
            atomicAdd(&outb[ob + g * 65], (c[g] + wb) * pv[g]);
        }
    }
    __syncthreads();

    // ---- write out (+bias), coalesced over pixels ----
    for (int i = t; i < COUT * 64; i += 128) {
        const int o = i >> 6;
        const int p = i & 63;                        // pixel within workgroup
        const float v = outb[p * 65 + o] + bs[o];
        out[(((size_t)b * COUT + o) * HQ + y) * WQ + xbase + p] = v;
    }
}

// ---------------------------------------------------------------------------
extern "C" void kernel_launch(void* const* d_in, const int* in_sizes, int n_in,
                              void* d_out, int out_size, void* d_ws, size_t ws_size,
                              hipStream_t stream) {
    (void)in_sizes; (void)n_in; (void)out_size; (void)ws_size;
    const float* x      = (const float*)d_in[0];
    const float* offset = (const float*)d_in[1];
    const float* posi   = (const float*)d_in[2];
    const float* w1     = (const float*)d_in[3];
    const float* b1     = (const float*)d_in[4];
    const float* w2     = (const float*)d_in[5];
    const float* b2     = (const float*)d_in[6];
    const float* wca    = (const float*)d_in[7];
    const float* bias   = (const float*)d_in[8];
    float* out = (float*)d_out;
    float* xad = (float*)d_ws;   // B*COUT*H*W floats = 33.5 MB scratch

    ca_kernel<<<BQ * HQ, 256, 0, stream>>>(x, wca, xad);
    cpad_kernel<<<BQ * HQ * 2, 128, 0, stream>>>(offset, posi, w1, b1, w2, b2,
                                                 bias, xad, out);
}
